// Decoder_88983132439257
// MI455X (gfx1250) — matrix operand order repaired and checked by reading, not compiled, not run on hardware
//
#include <hip/hip_runtime.h>
#include <hip/hip_bf16.h>

typedef __attribute__((ext_vector_type(16))) _Float16 v16h;
typedef __attribute__((ext_vector_type(8)))  _Float16 v8h;
typedef __attribute__((ext_vector_type(2)))  _Float16 v2h;
typedef __attribute__((ext_vector_type(8)))  float    v8f;

#define G_    8
#define NPG_  4096
#define HC_   16
#define LAT_  64
#define FF_   512
#define EDIM_ 3
#define EH_   64
#define NN_   (G_*NPG_)      // 32768 nodes
#define NE_   393216         // edges

// ---------------------------------------------------------------------------
// fc2: t[g,c] = x[g,:] @ fc2_w[:,c] + b[c]     (8x64 @ 64x512, one block)
// ---------------------------------------------------------------------------
__global__ __launch_bounds__(512) void fc2_kernel(const float* __restrict__ x,
    const float* __restrict__ w, const float* __restrict__ b, float* __restrict__ t)
{
    __shared__ float sx[G_*LAT_];          // 512 floats
    sx[threadIdx.x] = x[threadIdx.x];
    __syncthreads();
    const int c = threadIdx.x;
    float acc[G_];
    const float bb = b[c];
#pragma unroll
    for (int g = 0; g < G_; ++g) acc[g] = bb;
    for (int k = 0; k < LAT_; ++k) {
        const float wv = w[k*FF_ + c];
#pragma unroll
        for (int g = 0; g < G_; ++g) acc[g] = fmaf(sx[g*LAT_ + k], wv, acc[g]);
    }
#pragma unroll
    for (int g = 0; g < G_; ++g) t[g*FF_ + c] = acc[g];
}

// ---------------------------------------------------------------------------
// fc1: h0 = elu(t @ fc1_w + b)   (8x512 @ 512x65536) -> node features [N,16]
// Memory-bound on fc1_w (134 MB); thread-per-column, coalesced row reads.
// ---------------------------------------------------------------------------
__global__ __launch_bounds__(256) void fc1_kernel(const float* __restrict__ t,
    const float* __restrict__ w, const float* __restrict__ b, float* __restrict__ h0)
{
    __shared__ float st[G_*FF_];           // 16 KB
    for (int i = threadIdx.x; i < G_*FF_; i += 256) st[i] = t[i];
    __syncthreads();
    const int n = blockIdx.x*256 + threadIdx.x;   // output column 0..65535
    float acc[G_];
    const float bb = b[n];
#pragma unroll
    for (int g = 0; g < G_; ++g) acc[g] = bb;
    for (int k = 0; k < FF_; ++k) {
        const float wv = w[(size_t)k*(NPG_*HC_) + n];
#pragma unroll
        for (int g = 0; g < G_; ++g) acc[g] = fmaf(st[g*FF_ + k], wv, acc[g]);
    }
#pragma unroll
    for (int g = 0; g < G_; ++g) {
        float v = acc[g];
        h0[(size_t)g*(NPG_*HC_) + n] = v > 0.0f ? v : (__expf(v) - 1.0f);
    }
}

// ---------------------------------------------------------------------------
// Transpose + f16-convert w2 [64,256] -> w2T [256,64] for both conv layers.
// ---------------------------------------------------------------------------
__global__ __launch_bounds__(256) void prep_w2t_kernel(const float* __restrict__ w2a,
    const float* __restrict__ w2b, _Float16* __restrict__ ta, _Float16* __restrict__ tb)
{
    const int id = blockIdx.x*256 + threadIdx.x;  // 0..16383
    const int n = id >> 6, k = id & 63;
    ta[id] = (_Float16)w2a[k*256 + n];
    tb[id] = (_Float16)w2b[k*256 + n];
}

__global__ __launch_bounds__(256) void zero_kernel(float* __restrict__ p, int n)
{
    const int i = blockIdx.x*256 + threadIdx.x;
    if (i < n) p[i] = 0.0f;
}

// ---------------------------------------------------------------------------
// Fused NNConv edge kernel. Per wave: 16 edges.
//   h1 = relu(ea @ w1 + b1)           -> LDS f16, A-fragment layout
//   theta tile t = h1 @ w2T[t] + b2   -> v_wmma_f32_16x16x32_f16 (x2 over K=64)
//   msg[e][o] += x_src[e][t] * C_t[e][o]   (theta never leaves VGPRs)
//   atomicAdd scatter of msg into aggr[dst]
// ---------------------------------------------------------------------------
__global__ __launch_bounds__(256) void edge_conv_kernel(
    const float* __restrict__ h_in,       // [N,16]
    const int*   __restrict__ edge_index, // [2,NE]
    const float* __restrict__ edge_attr,  // [NE,3]
    const float* __restrict__ w1,         // [3,64]
    const float* __restrict__ b1,         // [64]
    const _Float16* __restrict__ w2T,     // [256,64] f16 (pre-transposed)
    const float* __restrict__ b2,         // [256]
    float* __restrict__ aggr)             // [N,16] atomic accumulation
{
    __shared__ _Float16 s_w2T[256*64];    // 32 KB
    __shared__ float s_w1[EDIM_*EH_];
    __shared__ float s_b1[EH_];
    __shared__ float s_b2[256];
    __shared__ _Float16 s_h1[8*16*72];    // per-wave A staging, stride 72 (16B aligned, conflict-free)
    __shared__ float s_xsT[8*16*16];      // per-wave gathered x[src], TRANSPOSED: [t][e]
    __shared__ int   s_src[8*16];
    __shared__ int   s_dst[8*16];

    const int tid = threadIdx.x;
    for (int i = tid; i < (256*64)/8; i += 256)
        ((v8h*)s_w2T)[i] = ((const v8h*)w2T)[i];
    for (int i = tid; i < EDIM_*EH_; i += 256) s_w1[i] = w1[i];
    if (tid < EH_)  s_b1[tid] = b1[tid];
    if (tid < 256)  s_b2[tid] = b2[tid];
    __syncthreads();

    const int wave = tid >> 5;
    const int lane = tid & 31;
    const int hi   = lane >> 4;   // half-wave select
    const int ln   = lane & 15;

    _Float16* my_h1  = s_h1  + wave*(16*72);
    float*    my_xsT = s_xsT + wave*(16*16);
    int*      my_src = s_src + wave*16;
    int*      my_dst = s_dst + wave*16;

    const int gwave  = blockIdx.x*8 + wave;
    const int nwaves = gridDim.x*8;
    const int ntiles = NE_/16;            // 24576

    for (int tile = gwave; tile < ntiles; tile += nwaves) {
        const int e0 = tile*16;
        // prefetch next tile's edge data (wave-uniform guard)
        if (tile + nwaves < ntiles) {
            const int ne0 = (tile + nwaves)*16;
            __builtin_prefetch(edge_attr + (size_t)ne0*EDIM_, 0, 3);
            __builtin_prefetch(edge_index + ne0, 0, 3);
            __builtin_prefetch(edge_index + NE_ + ne0, 0, 3);
        }
        if (lane < 16) {
            my_src[lane] = edge_index[e0 + lane];
            my_dst[lane] = edge_index[NE_ + e0 + lane];
        }
        // --- edge MLP layer 1: relu(ea@w1+b1) -> f16 LDS (A-major) ---
        {
            const int er = ln;
            const int cb = hi*32;
            const float* eap = edge_attr + (size_t)(e0 + er)*EDIM_;
            const float a0 = eap[0], a1v = eap[1], a2 = eap[2];
            v2h hv[16];
#pragma unroll
            for (int j = 0; j < 32; j += 2) {
                const int c = cb + j;
                float v0 = fmaf(a0, s_w1[c],
                           fmaf(a1v, s_w1[EH_ + c],
                           fmaf(a2, s_w1[2*EH_ + c], s_b1[c])));
                float v1 = fmaf(a0, s_w1[c+1],
                           fmaf(a1v, s_w1[EH_ + c+1],
                           fmaf(a2, s_w1[2*EH_ + c+1], s_b1[c+1])));
                hv[j>>1] = __builtin_bit_cast(v2h,
                    __builtin_amdgcn_cvt_pkrtz(fmaxf(v0, 0.0f), fmaxf(v1, 0.0f)));
            }
            v8h* dp = (v8h*)(my_h1 + er*72 + cb);
            const v8h* sp = (const v8h*)hv;
            dp[0]=sp[0]; dp[1]=sp[1]; dp[2]=sp[2]; dp[3]=sp[3];
        }
        // --- gather x[src], store transposed: xsT[t][e] ---
        {
            const int eg = lane >> 1;
            const int tb = (lane & 1)*8;
            const int sv = my_src[eg];
            const float4* hp = (const float4*)(h_in + (size_t)sv*HC_ + tb);
            const float4 u0 = hp[0], u1 = hp[1];
            my_xsT[(tb+0)*16 + eg] = u0.x;
            my_xsT[(tb+1)*16 + eg] = u0.y;
            my_xsT[(tb+2)*16 + eg] = u0.z;
            my_xsT[(tb+3)*16 + eg] = u0.w;
            my_xsT[(tb+4)*16 + eg] = u1.x;
            my_xsT[(tb+5)*16 + eg] = u1.y;
            my_xsT[(tb+6)*16 + eg] = u1.z;
            my_xsT[(tb+7)*16 + eg] = u1.w;
        }
        // --- load A fragments (16-bit A 16x32 layout; K=64 -> 2 frags) ---
        const int acol = hi*8;            // lane<16: K{0-7,16-23}; lane>=16: K{8-15,24-31}
        v16h A0, A1;
        {
            const v8h p0 = *(const v8h*)(my_h1 + ln*72 +      acol);
            const v8h p1 = *(const v8h*)(my_h1 + ln*72 + 16 + acol);
            const v8h p2 = *(const v8h*)(my_h1 + ln*72 + 32 + acol);
            const v8h p3 = *(const v8h*)(my_h1 + ln*72 + 48 + acol);
#pragma unroll
            for (int i = 0; i < 8; ++i) {
                A0[i] = p0[i]; A0[8+i] = p1[i];
                A1[i] = p2[i]; A1[8+i] = p3[i];
            }
        }
        float msg[8];
#pragma unroll
        for (int r = 0; r < 8; ++r) msg[r] = 0.0f;
        const int ebase = hi*8;

        // --- 16 output tiles: theta tile t = theta[:, i=t, :] ---
        // unroll 4: enough ILP to cover WMMA->VALU hazards without the
        // register blowup / scratch spills of a full unroll.
#pragma unroll 4
        for (int t = 0; t < 16; ++t) {
            const float cinit = s_b2[t*16 + ln];  // broadcast ds_load, fold b2 into C
            v8f C;
#pragma unroll
            for (int r = 0; r < 8; ++r) C[r] = cinit;
            const _Float16* wp = s_w2T + (t*16 + ln)*64 + hi*8;
            const v16h B0 = __builtin_shufflevector(*(const v8h*)(wp), *(const v8h*)(wp + 16), 0,1,2,3,4,5,6,7,8,9,10,11,12,13,14,15);           // K 0..31 half (table order)
            const v16h B1 = __builtin_shufflevector(*(const v8h*)(wp + 32), *(const v8h*)(wp + 48), 0,1,2,3,4,5,6,7,8,9,10,11,12,13,14,15);      // K 32..63 half (table order)
            C = __builtin_amdgcn_wmma_f32_16x16x32_f16(false, A0, false, B0, (short)0, C, false, false);
            C = __builtin_amdgcn_wmma_f32_16x16x32_f16(false, A1, false, B1, (short)0, C, false, false);
            // msg[e][o] += x_src[e][t] * theta[e][t][o]
            // xsT row: 8 consecutive floats -> 2x ds_load_b128, pairs feed v_pk_fma_f32
            const v8f xs = *(const v8f*)(my_xsT + t*16 + ebase);
#pragma unroll
            for (int r = 0; r < 8; ++r)
                msg[r] = fmaf(xs[r], C[r], msg[r]);
        }
        // --- scatter-add into aggr[dst] ---
#pragma unroll
        for (int r = 0; r < 8; ++r) {
            const int d = my_dst[ebase + r];
            atomicAdd(aggr + (size_t)d*HC_ + ln, msg[r]);
        }
    }
}

// ---------------------------------------------------------------------------
// out = elu(h @ root + aggr + bias)
// ---------------------------------------------------------------------------
__global__ __launch_bounds__(256) void node_update_kernel(
    const float* __restrict__ h, const float* __restrict__ aggr,
    const float* __restrict__ root, const float* __restrict__ bias,
    float* __restrict__ out)
{
    __shared__ float s_root[256];
    __shared__ float s_bias[16];
    s_root[threadIdx.x] = root[threadIdx.x];
    if (threadIdx.x < 16) s_bias[threadIdx.x] = bias[threadIdx.x];
    __syncthreads();
    const int gid = blockIdx.x*256 + threadIdx.x;    // n*16 + c
    const int n = gid >> 4, c = gid & 15;
    float acc = aggr[gid] + s_bias[c];
    const float* hr = h + (size_t)n*HC_;
#pragma unroll
    for (int k = 0; k < HC_; ++k) acc = fmaf(hr[k], s_root[k*16 + c], acc);
    out[gid] = acc > 0.0f ? acc : (__expf(acc) - 1.0f);
}

// ---------------------------------------------------------------------------
extern "C" void kernel_launch(void* const* d_in, const int* in_sizes, int n_in,
                              void* d_out, int out_size, void* d_ws, size_t ws_size,
                              hipStream_t stream)
{
    (void)in_sizes; (void)n_in; (void)out_size; (void)ws_size;
    const float* x     = (const float*)d_in[0];
    const int*   ei    = (const int*)d_in[1];
    const float* ea    = (const float*)d_in[2];
    const float* fc2_w = (const float*)d_in[3];
    const float* fc2_b = (const float*)d_in[4];
    const float* fc1_w = (const float*)d_in[5];
    const float* fc1_b = (const float*)d_in[6];
    const float* c_root[2] = {(const float*)d_in[7],  (const float*)d_in[13]};
    const float* c_bias[2] = {(const float*)d_in[8],  (const float*)d_in[14]};
    const float* c_w1[2]   = {(const float*)d_in[9],  (const float*)d_in[15]};
    const float* c_b1[2]   = {(const float*)d_in[10], (const float*)d_in[16]};
    const float* c_w2[2]   = {(const float*)d_in[11], (const float*)d_in[17]};
    const float* c_b2[2]   = {(const float*)d_in[12], (const float*)d_in[18]};

    char* ws = (char*)d_ws;
    float*    t_buf = (float*)(ws);                            // 16 KB
    float*    h0    = (float*)(ws + (16<<10));                 // 2 MB
    float*    h1n   = (float*)(ws + (16<<10) + (2u<<20));      // 2 MB
    float*    aggr  = (float*)(ws + (16<<10) + (4u<<20));      // 2 MB
    _Float16* w2T0  = (_Float16*)(ws + (16<<10) + (6u<<20));   // 32 KB
    _Float16* w2T1  = w2T0 + 256*64;                           // 32 KB

    fc2_kernel<<<1, 512, 0, stream>>>(x, fc2_w, fc2_b, t_buf);
    fc1_kernel<<<(NPG_*HC_)/256, 256, 0, stream>>>(t_buf, fc1_w, fc1_b, h0);
    prep_w2t_kernel<<<64, 256, 0, stream>>>(c_w2[0], c_w2[1], w2T0, w2T1);

    const float* hin = h0;
    float* houts[2] = { h1n, (float*)d_out };
    const _Float16* w2Ts[2] = { w2T0, w2T1 };
    for (int l = 0; l < 2; ++l) {
        zero_kernel<<<(NN_*HC_)/256, 256, 0, stream>>>(aggr, NN_*HC_);
        edge_conv_kernel<<<768, 256, 0, stream>>>(hin, ei, ea, c_w1[l], c_b1[l],
                                                  w2Ts[l], c_b2[l], aggr);
        node_update_kernel<<<(NN_*HC_)/256, 256, 0, stream>>>(hin, aggr, c_root[l],
                                                              c_bias[l], houts[l]);
        hin = houts[l];
    }
}
